// Sage_4672924418645
// MI455X (gfx1250) — matrix operand order repaired and checked by reading, not compiled, not run on hardware
//
#include <hip/hip_runtime.h>

typedef __attribute__((ext_vector_type(16))) _Float16 v16h;
typedef __attribute__((ext_vector_type(8)))  float    v8f;

#define N_NODES   100000
#define N_EDGES   1600000
#define IN_DIM    64
#define HIDDEN    64
#define N_CLASSES 249
#define KCAT      128   // concatenated K: 64 (aggregated) + 64 (self)

// ---------------------------------------------------------------------------
// Zero workspace (agg1 | h | agg2), float4 grid-stride stores.
// ---------------------------------------------------------------------------
__global__ void zero_ws(float4* __restrict__ p, long n4) {
  long i = (long)blockIdx.x * blockDim.x + threadIdx.x;
  long stride = (long)gridDim.x * blockDim.x;
  float4 z = {0.f, 0.f, 0.f, 0.f};
  for (; i < n4; i += stride) p[i] = z;
}

// ---------------------------------------------------------------------------
// Edge scatter-add: 32 lanes per edge, each lane handles features l and l+32.
// Gathers hit L2-resident feature matrix; atomics are f32 adds into agg.
// ---------------------------------------------------------------------------
__global__ void scatter_add64(const int* __restrict__ src,
                              const int* __restrict__ dst,
                              const float* __restrict__ feat,
                              float* __restrict__ agg,
                              long total) {
  long gid = (long)blockIdx.x * blockDim.x + threadIdx.x;
  if (gid >= total) return;
  long e = gid >> 5;
  int  l = (int)(gid & 31);
  long s = (long)src[e] * 64 + l;
  long d = (long)dst[e] * 64 + l;
  unsafeAtomicAdd(&agg[d],      feat[s]);
  unsafeAtomicAdd(&agg[d + 32], feat[s + 32]);
}

// Convert 8 consecutive f32 (16B-aligned) into 8 f16 lanes of an A fragment.
__device__ __forceinline__ void cvt8(const float* __restrict__ p, v16h& a, int base) {
  float4 u = ((const float4*)p)[0];
  float4 v = ((const float4*)p)[1];
  a[base + 0] = (_Float16)u.x; a[base + 1] = (_Float16)u.y;
  a[base + 2] = (_Float16)u.z; a[base + 3] = (_Float16)u.w;
  a[base + 4] = (_Float16)v.x; a[base + 5] = (_Float16)v.y;
  a[base + 6] = (_Float16)v.z; a[base + 7] = (_Float16)v.w;
}

// ---------------------------------------------------------------------------
// Layer 1: h = relu([agg1 | x] @ [W1l | W1r]^T + b1)   -> [N, 64]
// Block = 256 threads = 8 waves; each wave computes 16 rows x 64 cols.
// Weights staged in LDS as f16, row o = [W1l[o,:] | W1r[o,:]] (k-contiguous).
// ---------------------------------------------------------------------------
__global__ void __launch_bounds__(256)
sage_gemm1(const float* __restrict__ agg,   // [N,64]
           const float* __restrict__ x,     // [N,64]
           const float* __restrict__ W1l,   // [64,64]
           const float* __restrict__ W1r,   // [64,64]
           const float* __restrict__ b1,    // [64]
           float* __restrict__ hout)        // [N,64]
{
  __shared__ __align__(32) _Float16 Wt[HIDDEN * KCAT];   // 16 KB
  int tid = threadIdx.x;
  for (int i = tid; i < HIDDEN * KCAT; i += 256) {
    int o = i >> 7, k = i & 127;
    float w = (k < 64) ? W1l[o * 64 + k] : W1r[o * 64 + (k - 64)];
    Wt[i] = (_Float16)w;
  }
  __syncthreads();

  int wave = tid >> 5;
  int lane = tid & 31;
  int m    = lane & 15;
  int hi   = lane >> 4;                       // 0: lanes 0-15, 1: lanes 16-31
  long rowBase = (long)blockIdx.x * 128 + wave * 16;
  if (rowBase >= N_NODES) return;             // wave-uniform: EXEC stays all-1s
  long row = rowBase + m;

  v8f acc[4] = {v8f{}, v8f{}, v8f{}, v8f{}};
  int abase = hi * 16;                        // A layout half-wave K offset
  int kb    = hi * 16;                        // B layout half-wave K offset

#pragma unroll
  for (int kc = 0; kc < 4; ++kc) {
    const float* sp = (kc < 2) ? (agg + row * 64 + kc * 32 + abase)
                               : (x   + row * 64 + (kc - 2) * 32 + abase);
    v16h a;
    cvt8(sp,      a, 0);                      // K = 32*kc + abase      .. +7
    cvt8(sp + 8,  a, 8);                      // K = 32*kc + abase + 8  .. +15
    int koff = kc * 32 + kb;
#pragma unroll
    for (int c = 0; c < 4; ++c) {
      v16h b = *(const v16h*)&Wt[(c * 16 + m) * KCAT + koff];
      acc[c] = __builtin_amdgcn_wmma_f32_16x16x32_f16(
          false, a, false, b, (short)0, acc[c], false, false);
    }
  }

#pragma unroll
  for (int c = 0; c < 4; ++c) {
    int col = c * 16 + m;
    float bias = b1[col];
#pragma unroll
    for (int i = 0; i < 8; ++i) {
      long r = rowBase + hi * 8 + i;
      float v = acc[c][i] + bias;
      hout[r * 64 + col] = v > 0.f ? v : 0.f;
    }
  }
}

// ---------------------------------------------------------------------------
// Layer 2: out = [agg2 | h] @ [W2l | W2r]^T + b2   -> [N, 249] (cols padded to 256)
// Block = 256 threads = 8 waves laid out 2 (row groups) x 4 (col groups);
// each wave: 16 rows x 64 cols. Weights in LDS: 256 x 128 f16 = 64 KB.
// ---------------------------------------------------------------------------
__global__ void __launch_bounds__(256)
sage_gemm2(const float* __restrict__ agg2,  // [N,64]
           const float* __restrict__ hin,   // [N,64]
           const float* __restrict__ W2l,   // [249,64]
           const float* __restrict__ W2r,   // [249,64]
           const float* __restrict__ b2,    // [249]
           float* __restrict__ out)         // [N,249]
{
  __shared__ __align__(32) _Float16 Wt[256 * KCAT];      // 64 KB
  int tid = threadIdx.x;
  for (int i = tid; i < 256 * KCAT; i += 256) {
    int o = i >> 7, k = i & 127;
    float w = 0.f;
    if (o < N_CLASSES) w = (k < 64) ? W2l[o * 64 + k] : W2r[o * 64 + (k - 64)];
    Wt[i] = (_Float16)w;
  }
  __syncthreads();

  int wave = tid >> 5;
  int lane = tid & 31;
  int m    = lane & 15;
  int hi   = lane >> 4;
  int rowGroup = wave >> 2;
  int colBase  = (wave & 3) * 64;
  long rowBase = (long)blockIdx.x * 32 + rowGroup * 16;  // grid = 3125 exact
  long row = rowBase + m;

  v8f acc[4] = {v8f{}, v8f{}, v8f{}, v8f{}};
  int abase = hi * 16;
  int kb    = hi * 16;

#pragma unroll
  for (int kc = 0; kc < 4; ++kc) {
    const float* sp = (kc < 2) ? (agg2 + row * 64 + kc * 32 + abase)
                               : (hin  + row * 64 + (kc - 2) * 32 + abase);
    v16h a;
    cvt8(sp,      a, 0);
    cvt8(sp + 8,  a, 8);
    int koff = kc * 32 + kb;
#pragma unroll
    for (int c = 0; c < 4; ++c) {
      v16h b = *(const v16h*)&Wt[(colBase + c * 16 + m) * KCAT + koff];
      acc[c] = __builtin_amdgcn_wmma_f32_16x16x32_f16(
          false, a, false, b, (short)0, acc[c], false, false);
    }
  }

#pragma unroll
  for (int c = 0; c < 4; ++c) {
    int col = colBase + c * 16 + m;
    if (col < N_CLASSES) {
      float bias = b2[col];
#pragma unroll
      for (int i = 0; i < 8; ++i) {
        long r = rowBase + hi * 8 + i;
        out[r * (long)N_CLASSES + col] = acc[c][i] + bias;
      }
    }
  }
}

// ---------------------------------------------------------------------------
extern "C" void kernel_launch(void* const* d_in, const int* in_sizes, int n_in,
                              void* d_out, int out_size, void* d_ws, size_t ws_size,
                              hipStream_t stream) {
  const float* x    = (const float*)d_in[0];
  const int*   edge = (const int*)  d_in[1];   // [2, E] int32
  const float* W1l  = (const float*)d_in[2];
  const float* b1   = (const float*)d_in[3];
  const float* W1r  = (const float*)d_in[4];
  const float* W2l  = (const float*)d_in[5];
  const float* b2   = (const float*)d_in[6];
  const float* W2r  = (const float*)d_in[7];
  float* out = (float*)d_out;

  float* ws   = (float*)d_ws;
  float* agg1 = ws;                                  // [N,64]
  float* h    = ws + (long)N_NODES * 64;             // [N,64]
  float* agg2 = ws + 2l * (long)N_NODES * 64;        // [N,64]

  const int* src = edge;
  const int* dst = edge + N_EDGES;

  // 1) zero agg buffers (and h, harmless) : 19.2M floats
  long n4 = (3l * N_NODES * 64) / 4;
  zero_ws<<<4096, 256, 0, stream>>>((float4*)ws, n4);

  // 2) layer-1 aggregation: agg1[dst] += x[src]
  long totalLanes = (long)N_EDGES * 32;
  int  sblocks = (int)((totalLanes + 255) / 256);
  scatter_add64<<<sblocks, 256, 0, stream>>>(src, dst, x, agg1, totalLanes);

  // 3) h = relu([agg1|x] @ W1cat + b1)
  sage_gemm1<<<(N_NODES + 127) / 128, 256, 0, stream>>>(agg1, x, W1l, W1r, b1, h);

  // 4) layer-2 aggregation: agg2[dst] += h[src]
  scatter_add64<<<sblocks, 256, 0, stream>>>(src, dst, h, agg2, totalLanes);

  // 5) out = [agg2|h] @ W2cat + b2
  sage_gemm2<<<N_NODES / 32, 256, 0, stream>>>(agg2, h, W2l, W2r, b2, out);
}
